// MultiHeadAttention_22617297780971
// MI455X (gfx1250) — compile-verified, matrix operand order checked by reading, not run on hardware
//
#include <hip/hip_runtime.h>
#include <hip/hip_bf16.h>

// ---------------------------------------------------------------------------
// MHA forward for MI455X (gfx1250), wave32 + v_wmma_f32_16x16x32_bf16 and
// async global->LDS copies (ASYNCcnt) for double-buffered GEMM staging.
// Pipeline: cvt(fp32->bf16) -> QKV GEMM (WMMA, scatter per-head, scale folded
// into Q) -> flash attention (WMMA QK^T and PV, online softmax) -> proj GEMM.
// Workspace requirement: ~92.3 MB in d_ws.
// ---------------------------------------------------------------------------

typedef __bf16 bf16;
typedef __attribute__((ext_vector_type(16))) __bf16 v16bf;
typedef __attribute__((ext_vector_type(8)))  __bf16 v8bf;
typedef __attribute__((ext_vector_type(8)))  float  v8f;

#define EMBED 1024
#define HEADS 16
#define HD    64
#define SEQ   2048
#define BATCH 4
#define LOG2E 1.4426950408889634f
// softmax scale (1/sqrt(64)) * log2(e), folded into Q so we can use exp2f
#define QK_SCALE (0.125f * LOG2E)

__device__ __forceinline__ bf16 f2bf(float f) {
  union { float f; unsigned u; } v; v.f = f;
  unsigned r = (v.u + 0x7FFFu + ((v.u >> 16) & 1u)) >> 16;
  union { unsigned short s; bf16 b; } o; o.s = (unsigned short)r;
  return o.b;
}

struct bf16x16 { uint4 lo, hi; };
// Build a 16-element bf16 A/B fragment from two contiguous 16B chunks in LDS.
// Per the CDNA5 16-bit layout: per lane, elements 0..7 are K=klo..klo+7 and
// elements 8..15 are K=klo+16..klo+23 (klo = 0 for lanes 0-15, 8 for 16-31).
__device__ __forceinline__ v16bf load_frag(const bf16* p0, const bf16* p1) {
  bf16x16 t;
  t.lo = *(const uint4*)p0;
  t.hi = *(const uint4*)p1;
  return __builtin_bit_cast(v16bf, t);
}

__device__ __forceinline__ v8f wmma_bf16(v16bf a, v16bf b, v8f c) {
  return __builtin_amdgcn_wmma_f32_16x16x32_bf16(false, a, false, b,
                                                 (short)0, c, false, false);
}

// Async global->LDS 16B copy (per lane), tracked with ASYNCcnt.
// VDST operand = per-lane LDS byte address (low 32 bits of the generic ptr).
__device__ __forceinline__ void async_copy16(const bf16* g, bf16* l) {
  unsigned lds = (unsigned)(uintptr_t)l;
  asm volatile("global_load_async_to_lds_b128 %0, %1, off"
               :: "v"(lds), "v"(g) : "memory");
}
#define WAIT_ASYNC(N) asm volatile("s_wait_asynccnt " #N ::: "memory")

// ---------------------------------------------------------------------------
__global__ void cvt_f32_bf16(const float* __restrict__ src,
                             bf16* __restrict__ dst, int n) {
  int i = blockIdx.x * blockDim.x + threadIdx.x;
  int stride = gridDim.x * blockDim.x;
  for (; i < n; i += stride) dst[i] = f2bf(src[i]);
}

// ---------------------------------------------------------------------------
// C[M,Ncols] = A[M,K] * W[Ncols,K]^T + bias.  128x128 block tile, K-step 32,
// 256 threads = 8 waves, each wave a 64x32 sub-tile (4x2 wmma accum tiles).
// Double-buffered LDS staging via async global->LDS copies.
// mode 0: QKV epilogue -> bf16 scatter into Qb/Kb/Vb ([B*H][N][64]); Q scaled.
// mode 1: fp32 epilogue -> Out (row-major, Ncols columns).
__global__ __launch_bounds__(256)
void gemm_bf16_wmma(const bf16* __restrict__ A, const bf16* __restrict__ W,
                    const float* __restrict__ bias, int K, int mode,
                    bf16* __restrict__ Qb, bf16* __restrict__ Kb,
                    bf16* __restrict__ Vb, float* __restrict__ Out, int Ncols) {
  __shared__ bf16 As[2][128][48];   // M-major, 96B row stride (16B multiple)
  __shared__ bf16 Bs[2][128][48];   // N-major (rows of W), contiguous K

  const int tid  = threadIdx.x;
  const int lane = tid & 31;
  const int wid  = tid >> 5;
  const int wm   = (wid & 1) * 64;
  const int wn   = (wid >> 1) * 32;
  const int m0   = blockIdx.y * 128;
  const int n0   = blockIdx.x * 128;
  const int l15  = lane & 15;
  const int klo  = (lane >> 4) << 3;

  v8f acc[4][2];
#pragma unroll
  for (int i = 0; i < 4; ++i)
#pragma unroll
    for (int j = 0; j < 2; ++j)
#pragma unroll
      for (int e = 0; e < 8; ++e) acc[i][j][e] = 0.f;

  const int lrow = tid >> 1;
  const int lcol = (tid & 1) * 16;
  const bf16* gA = A + (size_t)(m0 + lrow) * K + lcol;  // + k offset per tile
  const bf16* gB = W + (size_t)(n0 + lrow) * K + lcol;

  // issue the 4 async 16B copies (per thread) for K-tile starting at kk
  auto issue_tile = [&](int kk, int buf) {
    async_copy16(gA + kk,     &As[buf][lrow][lcol]);
    async_copy16(gA + kk + 8, &As[buf][lrow][lcol + 8]);
    async_copy16(gB + kk,     &Bs[buf][lrow][lcol]);
    async_copy16(gB + kk + 8, &Bs[buf][lrow][lcol + 8]);
  };

  const int nk = K >> 5;
  issue_tile(0, 0);

  for (int t = 0; t < nk; ++t) {
    const int buf = t & 1;
    if (t + 1 < nk) {
      // prefetch next tile into the other buffer; safe: all waves passed the
      // trailing barrier of iter t-1, so nobody still reads buf^1.
      issue_tile((t + 1) << 5, buf ^ 1);
      WAIT_ASYNC(0x4);  // in-order completion: only the newest 4 may remain
    } else {
      WAIT_ASYNC(0x0);
    }
    __syncthreads();    // all waves' portions of tile t are in LDS

    v16bf af[4], bfg[2];
#pragma unroll
    for (int i = 0; i < 4; ++i) {
      const bf16* r = &As[buf][wm + i * 16 + l15][0];
      af[i] = load_frag(r + klo, r + klo + 16);
    }
#pragma unroll
    for (int j = 0; j < 2; ++j) {
      const bf16* r = &Bs[buf][wn + j * 16 + l15][0];
      bfg[j] = load_frag(r + klo, r + klo + 16);
    }
#pragma unroll
    for (int i = 0; i < 4; ++i)
#pragma unroll
      for (int j = 0; j < 2; ++j)
        acc[i][j] = wmma_bf16(af[i], bfg[j], acc[i][j]);
    __syncthreads();    // done reading buf before it gets overwritten
  }

  // Epilogue. C/D layout: element e -> row (e + 8*half), col = lane&15.
  const int half8 = (lane >> 4) * 8;
#pragma unroll
  for (int i = 0; i < 4; ++i)
#pragma unroll
    for (int j = 0; j < 2; ++j)
#pragma unroll
      for (int e = 0; e < 8; ++e) {
        int row = m0 + wm + i * 16 + half8 + e;
        int col = n0 + wn + j * 16 + l15;
        float v = acc[i][j][e] + bias[col];
        if (mode == 0) {
          int which = col >> 10;       // 0=q 1=k 2=v
          int h = (col & 1023) >> 6;
          int d = col & 63;
          int b = row >> 11;
          int n = row & 2047;
          size_t idx = ((size_t)((b * HEADS + h) * SEQ + n)) * HD + d;
          if (which == 0)      Qb[idx] = f2bf(v * QK_SCALE);
          else if (which == 1) Kb[idx] = f2bf(v);
          else                 Vb[idx] = f2bf(v);
        } else {
          Out[(size_t)row * Ncols + col] = v;
        }
      }
}

// ---------------------------------------------------------------------------
// Flash attention. grid = (SEQ/128, B*H). 8 waves x 16 query rows.
__global__ __launch_bounds__(256)
void flash_attn_wmma(const bf16* __restrict__ Qb, const bf16* __restrict__ Kb,
                     const bf16* __restrict__ Vb, bf16* __restrict__ Ob) {
  __shared__ bf16 Qs[128][72];       // 144B row stride
  __shared__ bf16 Ks[128][72];
  __shared__ bf16 Vt[64][136];       // V transposed: Vt[d][key], 272B stride
  __shared__ bf16 Ps[8][16][48];     // per-wave P tile (C-layout -> A-layout)

  const int tid  = threadIdx.x;
  const int lane = tid & 31;
  const int w    = tid >> 5;
  const int l15  = lane & 15;
  const int half = lane >> 4;
  const int klo  = half << 3;
  const int half8 = half * 8;
  const int bh = blockIdx.y;
  const int q0 = blockIdx.x * 128;
  const size_t base = (size_t)bh * SEQ * HD;

  // stage Q block [128][64] via async copies
#pragma unroll
  for (int i = 0; i < 4; ++i) {
    int vv = tid + i * 256;          // 0..1023 vec8 chunks
    int r = vv >> 3, c = (vv & 7) * 8;
    async_copy16(Qb + base + (size_t)(q0 + r) * HD + c, &Qs[r][c]);
  }
  WAIT_ASYNC(0x0);
  __syncthreads();

  v16bf aQ[2];
  {
    const bf16* r = &Qs[w * 16 + l15][0];
    aQ[0] = load_frag(r + klo,      r + klo + 16);
    aQ[1] = load_frag(r + 32 + klo, r + 32 + klo + 16);
  }

  v8f Oacc[4];
  float mrow[8], lrow[8];
#pragma unroll
  for (int t = 0; t < 4; ++t)
#pragma unroll
    for (int e = 0; e < 8; ++e) Oacc[t][e] = 0.f;
#pragma unroll
  for (int e = 0; e < 8; ++e) { mrow[e] = -1e30f; lrow[e] = 0.f; }

  for (int kb = 0; kb < SEQ; kb += 128) {
    // stage K block (async) and transposed V block (lane scatter)
#pragma unroll
    for (int i = 0; i < 4; ++i) {
      int vv = tid + i * 256;
      int r = vv >> 3, c = (vv & 7) * 8;
      async_copy16(Kb + base + (size_t)(kb + r) * HD + c, &Ks[r][c]);
      uint4 vq = *(const uint4*)(Vb + base + (size_t)(kb + r) * HD + c);
      v8bf vb = __builtin_bit_cast(v8bf, vq);
#pragma unroll
      for (int j = 0; j < 8; ++j) Vt[c + j][r] = vb[j];
    }
    WAIT_ASYNC(0x0);
    __syncthreads();

#pragma unroll
    for (int sb = 0; sb < 4; ++sb) {
      const int c0 = sb * 32;
      // S = (scaled Q) * K^T for 32 keys: two 16-col tiles, two K-dim steps
      v16bf bK[2][2];
#pragma unroll
      for (int j = 0; j < 2; ++j) {
        const bf16* r = &Ks[c0 + j * 16 + l15][0];
        bK[j][0] = load_frag(r + klo,      r + klo + 16);
        bK[j][1] = load_frag(r + 32 + klo, r + 32 + klo + 16);
      }
      v8f s0, s1;
#pragma unroll
      for (int e = 0; e < 8; ++e) { s0[e] = 0.f; s1[e] = 0.f; }
      s0 = wmma_bf16(aQ[0], bK[0][0], s0);
      s0 = wmma_bf16(aQ[1], bK[0][1], s0);
      s1 = wmma_bf16(aQ[0], bK[1][0], s1);
      s1 = wmma_bf16(aQ[1], bK[1][1], s1);

      // online softmax: per-row max/sum via width-16 shuffles (wave halves)
      float p0[8], p1[8];
#pragma unroll
      for (int e = 0; e < 8; ++e) {
        float mx = fmaxf(s0[e], s1[e]);
        mx = fmaxf(mx, __shfl_xor(mx, 1, 16));
        mx = fmaxf(mx, __shfl_xor(mx, 2, 16));
        mx = fmaxf(mx, __shfl_xor(mx, 4, 16));
        mx = fmaxf(mx, __shfl_xor(mx, 8, 16));
        float mnew  = fmaxf(mrow[e], mx);
        float alpha = exp2f(mrow[e] - mnew);
        mrow[e] = mnew;
        p0[e] = exp2f(s0[e] - mnew);
        p1[e] = exp2f(s1[e] - mnew);
        float rs = p0[e] + p1[e];
        rs += __shfl_xor(rs, 1, 16);
        rs += __shfl_xor(rs, 2, 16);
        rs += __shfl_xor(rs, 4, 16);
        rs += __shfl_xor(rs, 8, 16);
        lrow[e] = lrow[e] * alpha + rs;
#pragma unroll
        for (int t = 0; t < 4; ++t) Oacc[t][e] *= alpha;
        // dump P tile (C-layout) to per-wave LDS region
        Ps[w][half8 + e][l15]      = f2bf(p0[e]);
        Ps[w][half8 + e][16 + l15] = f2bf(p1[e]);
      }
      // wave-private LDS bounce: wait for our own ds stores before re-reading
      asm volatile("s_wait_dscnt 0" ::: "memory");

      v16bf aP;
      {
        const bf16* r = &Ps[w][l15][0];
        aP = load_frag(r + klo, r + klo + 16);
      }
      // O += P * V  (V fragments from transposed LDS tile, contiguous reads)
#pragma unroll
      for (int t = 0; t < 4; ++t) {
        const bf16* r = &Vt[t * 16 + l15][0];
        v16bf bV = load_frag(r + c0 + klo, r + c0 + klo + 16);
        Oacc[t] = wmma_bf16(aP, bV, Oacc[t]);
      }
    }
    __syncthreads();
  }

  // normalize and write out as bf16 in [B][N][H*D] layout for the proj GEMM
  const int b = bh / HEADS, h = bh % HEADS;
#pragma unroll
  for (int e = 0; e < 8; ++e) {
    float inv = 1.f / lrow[e];
    int n = q0 + w * 16 + half8 + e;
    size_t rowoff = ((size_t)(b * SEQ + n)) * EMBED + h * HD;
#pragma unroll
    for (int t = 0; t < 4; ++t) {
      int d = t * 16 + l15;
      Ob[rowoff + d] = f2bf(Oacc[t][e] * inv);
    }
  }
}

// ---------------------------------------------------------------------------
extern "C" void kernel_launch(void* const* d_in, const int* in_sizes, int n_in,
                              void* d_out, int out_size, void* d_ws, size_t ws_size,
                              hipStream_t stream) {
  const float* x      = (const float*)d_in[0];   // [4,2048,1024]
  const float* W_qkv  = (const float*)d_in[1];   // [3072,1024]
  const float* b_qkv  = (const float*)d_in[2];   // [3072]
  const float* W_proj = (const float*)d_in[3];   // [1024,1024]
  const float* b_proj = (const float*)d_in[4];   // [1024]
  float* out = (float*)d_out;                    // [4,2048,1024] fp32

  const int BNC  = BATCH * SEQ * EMBED;          // 8,388,608
  const int WQKV = 3 * EMBED * EMBED;            // 3,145,728
  const int WPRJ = EMBED * EMBED;                // 1,048,576
  const int QKV1 = BATCH * HEADS * SEQ * HD;     // 8,388,608

  bf16* xb     = (bf16*)d_ws;
  bf16* wqkvb  = xb + BNC;
  bf16* wprojb = wqkvb + WQKV;
  bf16* Qb     = wprojb + WPRJ;
  bf16* Kb     = Qb + QKV1;
  bf16* Vb     = Kb + QKV1;
  bf16* attnb  = Vb + QKV1;                      // total ~92.3 MB

  // 1) fp32 -> bf16 conversions
  cvt_f32_bf16<<<2048, 256, 0, stream>>>(x, xb, BNC);
  cvt_f32_bf16<<<1024, 256, 0, stream>>>(W_qkv, wqkvb, WQKV);
  cvt_f32_bf16<<<512, 256, 0, stream>>>(W_proj, wprojb, WPRJ);

  // 2) QKV projection: [8192,1024] x [3072,1024]^T, scatter per head
  gemm_bf16_wmma<<<dim3(3 * EMBED / 128, BATCH * SEQ / 128), 256, 0, stream>>>(
      xb, wqkvb, b_qkv, EMBED, 0, Qb, Kb, Vb, nullptr, 0);

  // 3) flash attention per (b,h), 128-query blocks
  flash_attn_wmma<<<dim3(SEQ / 128, BATCH * HEADS), 256, 0, stream>>>(
      Qb, Kb, Vb, attnb);

  // 4) output projection: [8192,1024] x [1024,1024]^T + bias -> fp32 out
  gemm_bf16_wmma<<<dim3(EMBED / 128, BATCH * SEQ / 128), 256, 0, stream>>>(
      attnb, wprojb, b_proj, EMBED, 1, nullptr, nullptr, nullptr, out, EMBED);
}
